// MedicalVQADecoder_37649683317223
// MI455X (gfx1250) — compile-verified, matrix operand order checked by reading, not run on hardware
//
#include <hip/hip_runtime.h>
#include <hip/hip_bf16.h>

#define HID    768
#define GATES  3072
#define VOCAB  30000
#define BATCH  64
#define MAXLEN 32

typedef __bf16 bf16x16 __attribute__((ext_vector_type(16)));
typedef float  f32x8   __attribute__((ext_vector_type(8)));

union FragU { bf16x16 v; uint4 q[2]; };

__device__ __forceinline__ unsigned short f2bf(float f) {
  unsigned u = __float_as_uint(f);
  unsigned r = 0x7FFFu + ((u >> 16) & 1u);     // round-to-nearest-even
  return (unsigned short)((u + r) >> 16);
}

// Load one 16x32 bf16 fragment row for this lane: two contiguous 16B runs.
// lanes 0-15: K = k0+{0..7} and k0+{16..23}; lanes 16-31: K = k0+{8..15} and k0+{24..31}
__device__ __forceinline__ bf16x16 load_frag(const unsigned short* __restrict__ base,
                                             int row, int ld, int k0, int lane) {
  FragU f;
  int run0 = k0 + ((lane & 16) ? 8 : 0);
  f.q[0] = *reinterpret_cast<const uint4*>(base + row * ld + run0);
  f.q[1] = *reinterpret_cast<const uint4*>(base + row * ld + run0 + 16);
  return f.v;
}

__device__ __forceinline__ unsigned long long shfl_xor_u64(unsigned long long v, int mask) {
  unsigned lo = (unsigned)v, hi = (unsigned)(v >> 32);
  lo = __shfl_xor(lo, mask, 32);
  hi = __shfl_xor(hi, mask, 32);
  return ((unsigned long long)hi << 32) | lo;
}

// ---------------- one-time prep ----------------
__global__ __launch_bounds__(256) void convert_weights(
    const float* __restrict__ Wih, const float* __restrict__ Whh,
    const float* __restrict__ Wout,
    unsigned short* __restrict__ dWih, unsigned short* __restrict__ dWhh,
    unsigned short* __restrict__ dWout) {
  const int n1 = GATES * HID;            // 2,359,296
  const int n3 = VOCAB * HID;            // 23,040,000
  const int total = 2 * n1 + n3;         // 27,758,592
  for (int i = blockIdx.x * blockDim.x + threadIdx.x; i < total;
       i += gridDim.x * blockDim.x) {
    if (i < n1)            dWih[i]          = f2bf(Wih[i]);
    else if (i < 2 * n1)   dWhh[i - n1]     = f2bf(Whh[i - n1]);
    else                   dWout[i - 2*n1]  = f2bf(Wout[i - 2*n1]);
  }
}

__global__ __launch_bounds__(256) void init_state(
    const float* __restrict__ feat, const float* __restrict__ emb,
    unsigned short* __restrict__ h_bf, unsigned short* __restrict__ x_bf,
    float* __restrict__ c, unsigned long long* __restrict__ slots) {
  int idx = blockIdx.x * blockDim.x + threadIdx.x;   // 64*768
  int k = idx % HID;
  h_bf[idx] = f2bf(feat[idx]);        // h0 = feat
  x_bf[idx] = f2bf(emb[k]);           // x0 = emb[token 0]
  c[idx] = 0.0f;
  if (idx < BATCH) slots[idx] = 0ull;
}

// ---------------- closed-set head (exact fp32) ----------------
__global__ __launch_bounds__(256) void closed_head(
    const float* __restrict__ feat,
    const float* __restrict__ c1W, const float* __restrict__ c1b,
    const float* __restrict__ c2W, const float* __restrict__ c2b,
    float* __restrict__ out) {
  __shared__ float hid[512];
  __shared__ float red0[256], red1[256];
  int b = blockIdx.x, tid = threadIdx.x;
  const float* f = feat + b * HID;
  for (int j = tid; j < 512; j += 256) {
    const float* w = c1W + j * HID;
    float s = c1b[j];
    for (int k = 0; k < HID; ++k) s += f[k] * w[k];
    hid[j] = fmaxf(s, 0.0f);
  }
  __syncthreads();
  float p0 = 0.f, p1 = 0.f;
  for (int j = tid; j < 512; j += 256) {
    p0 += hid[j] * c2W[j];
    p1 += hid[j] * c2W[512 + j];
  }
  red0[tid] = p0; red1[tid] = p1;
  __syncthreads();
  for (int s = 128; s; s >>= 1) {
    if (tid < s) { red0[tid] += red0[tid + s]; red1[tid] += red1[tid + s]; }
    __syncthreads();
  }
  if (tid == 0) {
    out[b * 2 + 0] = red0[0] + c2b[0];
    out[b * 2 + 1] = red1[0] + c2b[1];
  }
}

// ---------------- gate GEMM: gates = x@Wih^T + h@Whh^T + b_ih + b_hh ----------------
__global__ __launch_bounds__(256) void gates_gemm(
    const unsigned short* __restrict__ x, const unsigned short* __restrict__ h,
    const unsigned short* __restrict__ Wih, const unsigned short* __restrict__ Whh,
    const float* __restrict__ b_ih, const float* __restrict__ b_hh,
    float* __restrict__ gates) {
  int tid = threadIdx.x, lane = tid & 31, wave = tid >> 5;
  int m0 = (wave & 3) * 16;
  int n0 = blockIdx.x * 32 + (wave >> 2) * 16;
  int mrow = m0 + (lane & 15);
  int nrow = n0 + (lane & 15);
  f32x8 acc = {};
  for (int k0 = 0; k0 < HID; k0 += 32) {
    bf16x16 a = load_frag(x,   mrow, HID, k0, lane);
    bf16x16 b = load_frag(Wih, nrow, HID, k0, lane);
    acc = __builtin_amdgcn_wmma_f32_16x16x32_bf16(false, a, false, b, (short)0, acc, false, false);
  }
  for (int k0 = 0; k0 < HID; k0 += 32) {
    bf16x16 a = load_frag(h,   mrow, HID, k0, lane);
    bf16x16 b = load_frag(Whh, nrow, HID, k0, lane);
    acc = __builtin_amdgcn_wmma_f32_16x16x32_bf16(false, a, false, b, (short)0, acc, false, false);
  }
  float bias = b_ih[nrow] + b_hh[nrow];
  int mbase = m0 + ((lane & 16) ? 8 : 0);
#pragma unroll
  for (int r = 0; r < 8; ++r)
    gates[(mbase + r) * GATES + nrow] = acc[r] + bias;
}

// ---------------- LSTM cell (elementwise fp32) ----------------
__global__ __launch_bounds__(256) void lstm_cell_k(
    const float* __restrict__ gates, float* __restrict__ c,
    unsigned short* __restrict__ h_bf) {
  int idx = blockIdx.x * blockDim.x + threadIdx.x;   // 64*768
  int b = idx / HID, j = idx % HID;
  const float* g = gates + b * GATES;
  float ig = g[j], fg = g[HID + j], gg = g[2 * HID + j], og = g[3 * HID + j];
  float si = 1.f / (1.f + __expf(-ig));
  float sf = 1.f / (1.f + __expf(-fg));
  float so = 1.f / (1.f + __expf(-og));
  float cn = sf * c[idx] + si * tanhf(gg);
  c[idx] = cn;
  h_bf[idx] = f2bf(so * tanhf(cn));
}

// ---------------- logits GEMM + fused row argmax ----------------
__global__ __launch_bounds__(256) void logits_argmax(
    const unsigned short* __restrict__ h, const unsigned short* __restrict__ Wout,
    const float* __restrict__ out_b, unsigned long long* __restrict__ slots) {
  int tid = threadIdx.x, lane = tid & 31, wave = tid >> 5;
  int m0 = (wave & 3) * 16;
  int n0 = blockIdx.x * 32 + (wave >> 2) * 16;
  int mrow = m0 + (lane & 15);
  int ncol = n0 + (lane & 15);
  bool valid = (ncol < VOCAB);
  f32x8 acc = {};
  for (int k0 = 0; k0 < HID; k0 += 32) {
    bf16x16 a = load_frag(h, mrow, HID, k0, lane);
    bf16x16 b;
    if (valid) {
      b = load_frag(Wout, ncol, HID, k0, lane);
      if (k0 + 32 < HID)
        __builtin_prefetch(Wout + ncol * HID + k0 + 32, 0, 3);
    } else {
      FragU z; z.q[0] = make_uint4(0,0,0,0); z.q[1] = make_uint4(0,0,0,0);
      b = z.v;
    }
    acc = __builtin_amdgcn_wmma_f32_16x16x32_bf16(false, a, false, b, (short)0, acc, false, false);
  }
  float bias = valid ? out_b[ncol] : 0.0f;
#pragma unroll
  for (int r = 0; r < 8; ++r) {
    float v = acc[r] + bias;
    unsigned u = __float_as_uint(v);
    unsigned s = (u & 0x80000000u) ? ~u : (u | 0x80000000u);  // order-preserving
    unsigned long long packed =
        valid ? (((unsigned long long)s << 32) | (unsigned)(~(unsigned)ncol)) : 0ull;
    // reduce within each 16-lane half (rows m0+r and m0+8+r)
#pragma unroll
    for (int off = 8; off; off >>= 1) {
      unsigned long long o = shfl_xor_u64(packed, off);
      if (o > packed) packed = o;
    }
    if ((lane & 15) == 0 && packed) {
      int m = m0 + r + ((lane & 16) ? 8 : 0);
      atomicMax(&slots[m], packed);
    }
  }
}

// ---------------- decode token, gather next embedding, emit output ----------------
__global__ __launch_bounds__(256) void finalize_step(
    unsigned long long* __restrict__ slots, const float* __restrict__ emb,
    unsigned short* __restrict__ x_bf, float* __restrict__ gen_out, int t) {
  int b = blockIdx.x;
  unsigned long long s = slots[b];
  int tok = (int)(~(unsigned)(s & 0xFFFFFFFFull));
  __syncthreads();                       // everyone has read the slot
  for (int k = threadIdx.x; k < HID; k += blockDim.x)
    x_bf[b * HID + k] = f2bf(emb[(long long)tok * HID + k]);
  if (threadIdx.x == 0) {
    gen_out[b * MAXLEN + t] = (float)tok;
    slots[b] = 0ull;                     // reset for next step
  }
}

extern "C" void kernel_launch(void* const* d_in, const int* in_sizes, int n_in,
                              void* d_out, int out_size, void* d_ws, size_t ws_size,
                              hipStream_t stream) {
  const float* feat  = (const float*)d_in[0];   // (64,1,768)
  const float* emb   = (const float*)d_in[1];   // (30000,768)
  const float* W_ih  = (const float*)d_in[2];   // (3072,768)
  const float* W_hh  = (const float*)d_in[3];   // (3072,768)
  const float* b_ih  = (const float*)d_in[4];
  const float* b_hh  = (const float*)d_in[5];
  const float* out_W = (const float*)d_in[6];   // (30000,768)
  const float* out_b = (const float*)d_in[7];
  const float* c1W   = (const float*)d_in[8];
  const float* c1b   = (const float*)d_in[9];
  const float* c2W   = (const float*)d_in[10];
  const float* c2b   = (const float*)d_in[11];
  float* out = (float*)d_out;                   // [0,128): closed logits, [128,2176): tokens

  char* ws = (char*)d_ws;
  unsigned short* dWih  = (unsigned short*)(ws);
  unsigned short* dWhh  = (unsigned short*)(ws + 4718592);
  unsigned short* dWout = (unsigned short*)(ws + 9437184);
  unsigned short* x_bf  = (unsigned short*)(ws + 55517184);
  unsigned short* h_bf  = (unsigned short*)(ws + 55615488);
  float*          cbuf  = (float*)(ws + 55713792);
  float*          gates = (float*)(ws + 55910400);
  unsigned long long* slots = (unsigned long long*)(ws + 56696832);

  convert_weights<<<2048, 256, 0, stream>>>(W_ih, W_hh, out_W, dWih, dWhh, dWout);
  init_state<<<(BATCH * HID) / 256, 256, 0, stream>>>(feat, emb, h_bf, x_bf, cbuf, slots);
  closed_head<<<BATCH, 256, 0, stream>>>(feat, c1W, c1b, c2W, c2b, out);

  const int gate_blocks  = GATES / 32;                 // 96
  const int logit_blocks = (VOCAB + 31) / 32;          // 938
  for (int t = 0; t < MAXLEN; ++t) {
    gates_gemm<<<gate_blocks, 256, 0, stream>>>(x_bf, h_bf, dWih, dWhh, b_ih, b_hh, gates);
    lstm_cell_k<<<(BATCH * HID) / 256, 256, 0, stream>>>(gates, cbuf, h_bf);
    logits_argmax<<<logit_blocks, 256, 0, stream>>>(h_bf, dWout, out_b, slots);
    finalize_step<<<BATCH, 256, 0, stream>>>(slots, emb, x_bf, out + BATCH * 2, t);
  }
}
